// UnrollGN_41601053229213
// MI455X (gfx1250) — matrix operand order repaired and checked by reading, not compiled, not run on hardware
//
#include <hip/hip_runtime.h>

#define HIDDEN   1024
#define SPW      32            // samples per workgroup
#define THREADS  512           // 16 wave32 waves
#define N_ITER   5
#define KSTEPS   (HIDDEN / 32) // 32 k-steps of 32
#define H1STRIDE 1028          // padded row stride (floats) to break LDS bank conflicts

typedef __attribute__((ext_vector_type(16))) _Float16 v16h;
typedef __attribute__((ext_vector_type(8)))  float    v8f;

// 3-product split-precision WMMA: (Ah+Al) x (Bh+Bl) ~= AhBh + AhBl + AlBh
__device__ __forceinline__ v8f wmma3(v16h ah, v16h al, v16h bh, v16h bl, v8f acc) {
  acc = __builtin_amdgcn_wmma_f32_16x16x32_f16(false, ah, false, bh, (short)0, acc, false, false);
  acc = __builtin_amdgcn_wmma_f32_16x16x32_f16(false, ah, false, bl, (short)0, acc, false, false);
  acc = __builtin_amdgcn_wmma_f32_16x16x32_f16(false, al, false, bh, (short)0, acc, false, false);
  return acc;
}

// Pre-pack W2 (f32 row-major [K=1024][N=1024]) into f16 hi/lo, swizzled into the
// WMMA B-fragment layout: tile (kb,nb) of shape 32(K)x16(N); lane l holds 16
// contiguous halves t=0..15 with K = kb*32 + (l<16 ? t : 16+t), N = nb*16 + (l&15).
__global__ void UnrollGN_pack_w2(const float* __restrict__ W2,
                                 _Float16* __restrict__ hi,
                                 _Float16* __restrict__ lo) {
  int idx = blockIdx.x * 256 + threadIdx.x;   // 0 .. 1024*1024-1
  int i = idx >> 10;                          // K index
  int j = idx & (HIDDEN - 1);                 // N index
  float w = W2[idx];
  _Float16 h = (_Float16)w;
  _Float16 l = (_Float16)(w - (float)h);
  int kb = i >> 5, koff = i & 31;
  int lanehalf = koff >> 4;                   // 0: lanes 0-15, 1: lanes 16-31
  int t  = koff & 15;
  int nb = j >> 4, nn = j & 15;
  int lane = (lanehalf << 4) | nn;
  size_t dst = ((size_t)(kb * 64 + nb)) * 512 + (size_t)lane * 16 + t;
  hi[dst] = h;
  lo[dst] = l;
}

__global__ __launch_bounds__(THREADS)
void UnrollGN_gn_kernel(const float* __restrict__ x,  const float* __restrict__ W1,
                        const float* __restrict__ b1, const float* __restrict__ b2,
                        const float* __restrict__ W3, const float* __restrict__ b3,
                        const _Float16* __restrict__ w2hi,
                        const _Float16* __restrict__ w2lo,
                        float* __restrict__ out) {
  __shared__ float lds_h1[SPW * H1STRIDE];           // 128.5 KB, padded rows
  __shared__ float lds_w1r0[HIDDEN], lds_w1r1[HIDDEN], lds_b1[HIDDEN];
  __shared__ float lds_b2[HIDDEN], lds_w3[HIDDEN];
  __shared__ float lds_x[SPW], lds_y[SPW], lds_E[SPW], lds_g[SPW];

  const int tid     = threadIdx.x;
  const int wave    = tid >> 5;
  const int lane    = tid & 31;
  const int m       = lane & 15;     // row within 16-row tile / N within tile
  const int halfsel = lane >> 4;     // K-half selector per WMMA A/B layout
  const int wg      = blockIdx.x;

  for (int i = tid; i < HIDDEN; i += THREADS) {
    lds_w1r0[i] = W1[i];
    lds_w1r1[i] = W1[HIDDEN + i];
    lds_b1[i]   = b1[i];
    lds_b2[i]   = b2[i];
    lds_w3[i]   = W3[i];
  }
  if (tid < SPW) {
    lds_x[tid] = x[wg * SPW + tid];
    lds_y[tid] = 0.0f;                      // init_scheme == 'zero'
  }
  const float b3v = b3[0];
  __syncthreads();

  for (int it = 0; it < N_ITER; ++it) {
    if (tid < SPW) { lds_E[tid] = 0.0f; lds_g[tid] = 0.0f; }
    // Stage h1 = tanh(x*W1[0] + y*W1[1] + b1) for all 32 samples.
    for (int idx = tid; idx < SPW * HIDDEN; idx += THREADS) {
      int s = idx >> 10, i = idx & (HIDDEN - 1);
      float pre = lds_x[s] * lds_w1r0[i] + lds_y[s] * lds_w1r1[i] + lds_b1[i];
      lds_h1[s * H1STRIDE + i] = tanhf(pre);
    }
    __syncthreads();

    // Accumulators: [mtile 0..1][ntile slot 0..3] for both a = h1@W2 and c = u@W2.
    v8f acc_a[2][4], acc_c[2][4];
    #pragma unroll
    for (int mt = 0; mt < 2; ++mt)
      #pragma unroll
      for (int i = 0; i < 4; ++i) {
        v8f z = {0.f, 0.f, 0.f, 0.f, 0.f, 0.f, 0.f, 0.f};
        acc_a[mt][i] = z;
        acc_c[mt][i] = z;
      }

    for (int kb = 0; kb < KSTEPS; ++kb) {
      const int kbase = kb * 32;
      // Contiguous A layout matching packed B: lanes 0-15 hold K = {0..15},
      // lanes 16-31 hold K = {16..31} (relative to kbase).
      const int r0 = kbase + (halfsel ? 16 : 0);
      const int r1 = kbase + (halfsel ? 24 : 8);

      v16h ah[2], al[2], uh[2], ul[2];
      #pragma unroll
      for (int mt = 0; mt < 2; ++mt) {
        const float* hrow = &lds_h1[(mt * 16 + m) * H1STRIDE];
        #pragma unroll
        for (int t = 0; t < 8; ++t) {
          float h0  = hrow[r0 + t];
          float h1v = hrow[r1 + t];
          float w0  = lds_w1r1[r0 + t];
          float w1v = lds_w1r1[r1 + t];
          float u0  = (1.0f - h0  * h0 ) * w0;   // dh1/dy
          float u1  = (1.0f - h1v * h1v) * w1v;
          _Float16 hh0 = (_Float16)h0;  ah[mt][t]     = hh0; al[mt][t]     = (_Float16)(h0  - (float)hh0);
          _Float16 hh1 = (_Float16)h1v; ah[mt][t + 8] = hh1; al[mt][t + 8] = (_Float16)(h1v - (float)hh1);
          _Float16 uh0 = (_Float16)u0;  uh[mt][t]     = uh0; ul[mt][t]     = (_Float16)(u0  - (float)uh0);
          _Float16 uh1 = (_Float16)u1;  uh[mt][t + 8] = uh1; ul[mt][t + 8] = (_Float16)(u1  - (float)uh1);
        }
      }

      #pragma unroll
      for (int i = 0; i < 4; ++i) {
        const int nb = wave + (i << 4);                     // ntile 0..63
        const size_t tb = ((size_t)(kb * 64 + nb)) * 512 + (size_t)lane * 16;
        v16h bh = *(const v16h*)(w2hi + tb);                // 32B aligned
        v16h bl = *(const v16h*)(w2lo + tb);
        #pragma unroll
        for (int mt = 0; mt < 2; ++mt) {
          acc_a[mt][i] = wmma3(ah[mt], al[mt], bh, bl, acc_a[mt][i]);
          acc_c[mt][i] = wmma3(uh[mt], ul[mt], bh, bl, acc_c[mt][i]);
        }
      }
    }

    // Epilogue straight from WMMA C layout: VGPR r -> row r (lanes 0-15) / r+8
    // (lanes 16-31); N = lane&15. Reduce over the 16 lanes sharing each row.
    #pragma unroll
    for (int i = 0; i < 4; ++i) {
      const int nb  = wave + (i << 4);
      const int j   = nb * 16 + m;
      const float b2j = lds_b2[j];
      const float w3j = lds_w3[j];
      #pragma unroll
      for (int mt = 0; mt < 2; ++mt) {
        #pragma unroll
        for (int r = 0; r < 8; ++r) {
          float av = acc_a[mt][i][r] + b2j;
          float h2 = tanhf(av);
          float e  = w3j * h2;                               // -> E
          float gg = w3j * (1.0f - h2 * h2) * acc_c[mt][i][r]; // -> dE/dy
          #pragma unroll
          for (int mask = 1; mask < 16; mask <<= 1) {        // reduce within 16-lane halves
            e  += __shfl_xor(e,  mask, 32);
            gg += __shfl_xor(gg, mask, 32);
          }
          if (m == 0) {
            int s = mt * 16 + r + (halfsel << 3);
            atomicAdd(&lds_E[s], e);
            atomicAdd(&lds_g[s], gg);
          }
        }
      }
    }
    __syncthreads();

    if (tid < SPW) {
      float E = lds_E[tid] + b3v;
      lds_y[tid] -= E / lds_g[tid];        // 1-D Gauss-Newton, INNER_LR = 1
    }
    __syncthreads();
  }

  if (tid < SPW) out[wg * SPW + tid] = lds_y[tid];
}

extern "C" void kernel_launch(void* const* d_in, const int* in_sizes, int n_in,
                              void* d_out, int out_size, void* d_ws, size_t ws_size,
                              hipStream_t stream) {
  (void)in_sizes; (void)n_in; (void)out_size; (void)ws_size;
  const float* x  = (const float*)d_in[0];
  const float* W1 = (const float*)d_in[1];
  const float* b1 = (const float*)d_in[2];
  const float* W2 = (const float*)d_in[3];
  const float* b2 = (const float*)d_in[4];
  const float* W3 = (const float*)d_in[5];
  const float* b3 = (const float*)d_in[6];
  float* out = (float*)d_out;

  _Float16* w2hi = (_Float16*)d_ws;                     // 2 MB
  _Float16* w2lo = w2hi + (size_t)HIDDEN * HIDDEN;      // 2 MB

  UnrollGN_pack_w2<<<(HIDDEN * HIDDEN) / 256, 256, 0, stream>>>(W2, w2hi, w2lo);
  UnrollGN_gn_kernel<<<16384 / SPW, THREADS, 0, stream>>>(x, W1, b1, b2, W3, b3,
                                                          w2hi, w2lo, out);
}
